// LSTMCell_8753143349928
// MI455X (gfx1250) — compile-verified, matrix operand order checked by reading, not run on hardware
//
#include <hip/hip_runtime.h>
#include <hip/hip_bf16.h>

// LSTM cell, B=8192, I=H=1024, fused dual-GEMM (K=2048) + gate nonlinearities.
// bf16 WMMA (v_wmma_f32_16x16x32_bf16), f32 accumulate, double-buffered LDS.

typedef __attribute__((ext_vector_type(16))) __bf16 v16bf;
typedef __attribute__((ext_vector_type(8)))  __bf16 v8bf;
typedef __attribute__((ext_vector_type(4)))  __bf16 v4bf;
typedef __attribute__((ext_vector_type(8)))  float  v8f;

#define B_DIM 8192
#define H_DIM 1024
#define KTOT  2048
#define LDSS  40   // LDS row stride in bf16 elements (80 B: 16B-aligned, staggers banks)

__device__ __forceinline__ float fast_sigmoid(float x) {
    return 1.0f / (1.0f + __expf(-x));
}
__device__ __forceinline__ float fast_tanh(float x) {
    float a = fabsf(x);
    float t = __expf(-2.0f * a);
    float r = (1.0f - t) / (1.0f + t);
    return copysignf(r, x);
}
__device__ __forceinline__ v4bf cvt4(float4 v) {
    v4bf r;
    r[0] = (__bf16)v.x; r[1] = (__bf16)v.y; r[2] = (__bf16)v.z; r[3] = (__bf16)v.w;
    return r;
}
// Per-lane WMMA 16-bit operand fragment from an LDS tile row (M x 32 or N x 32):
// lanes 0-15 hold K=0..7 then K=16..23; lanes 16-31 hold K=8..15 then K=24..31.
__device__ __forceinline__ v16bf ldsFrag(const __bf16* rowBase, int lane) {
    const int kh = (lane & 16) ? 8 : 0;
    v8bf lo = *(const v8bf*)(rowBase + kh);
    v8bf hi = *(const v8bf*)(rowBase + 16 + kh);
    return __builtin_shufflevector(lo, hi, 0,1,2,3,4,5,6,7,8,9,10,11,12,13,14,15);
}

__global__ __launch_bounds__(256) void lstm_wmma_kernel(
    const float* __restrict__ x,    const float* __restrict__ hx,
    const float* __restrict__ cx,
    const float* __restrict__ W_xh, const float* __restrict__ b_xh,
    const float* __restrict__ W_hh, const float* __restrict__ b_hh,
    float* __restrict__ out)
{
    // A tile: 128 rows (batch) x 32 K.  B tile: 4 gates x 64 cols x 32 K.
    __shared__ __bf16 As[2][128 * LDSS];
    __shared__ __bf16 Bs[2][256 * LDSS];

    const int tid  = threadIdx.x;
    const int lane = tid & 31;
    const int wid  = tid >> 5;
    const int wm   = wid & 3;          // wave M index 0..3  (32 rows each)
    const int wn   = wid >> 2;         // wave N index 0..1  (32 cols each)
    const int m0   = blockIdx.y * 128; // batch tile
    const int n0   = blockIdx.x * 64;  // column tile within H

    v8f acc[4][2][2];
    {
        v8f z = {0.f,0.f,0.f,0.f,0.f,0.f,0.f,0.f};
        #pragma unroll
        for (int g = 0; g < 4; ++g)
            #pragma unroll
            for (int mt = 0; mt < 2; ++mt)
                #pragma unroll
                for (int nt = 0; nt < 2; ++nt)
                    acc[g][mt][nt] = z;
    }

    float4 aReg[4];
    float4 bReg[8];

    auto loadRegs = [&](int kc) {
        const int kk = kc * 32;
        const float* Asrc; const float* Bsrc; int ko;
        if (kk < 1024) { Asrc = x;  Bsrc = W_xh; ko = kk; }
        else           { Asrc = hx; Bsrc = W_hh; ko = kk - 1024; }
        #pragma unroll
        for (int i = 0; i < 4; ++i) {       // 128 x 32 fp32, float4-wide
            int idx = tid + i * 256;
            int row = idx >> 3, c4 = idx & 7;
            aReg[i] = *(const float4*)(Asrc + (size_t)(m0 + row) * 1024 + ko + c4 * 4);
        }
        #pragma unroll
        for (int i = 0; i < 8; ++i) {       // (4*64) x 32 fp32
            int idx = tid + i * 256;
            int row = idx >> 3, c4 = idx & 7;
            int grow = (row >> 6) * 1024 + n0 + (row & 63); // gate*1024 + col
            bReg[i] = *(const float4*)(Bsrc + (size_t)grow * 1024 + ko + c4 * 4);
        }
    };
    auto storeLDS = [&](int buf) {
        #pragma unroll
        for (int i = 0; i < 4; ++i) {
            int idx = tid + i * 256;
            int row = idx >> 3, c4 = idx & 7;
            *(v4bf*)&As[buf][row * LDSS + c4 * 4] = cvt4(aReg[i]);
        }
        #pragma unroll
        for (int i = 0; i < 8; ++i) {
            int idx = tid + i * 256;
            int row = idx >> 3, c4 = idx & 7;
            *(v4bf*)&Bs[buf][row * LDSS + c4 * 4] = cvt4(bReg[i]);
        }
    };

    loadRegs(0);
    storeLDS(0);
    int cur = 0;
    const int NKC = KTOT / 32;   // 64 K-chunks
    for (int kc = 0; kc < NKC; ++kc) {
        __syncthreads();                        // buf `cur` ready; other buf free
        if (kc + 1 < NKC) loadRegs(kc + 1);     // global loads overlap WMMA

        v16bf afr[2];
        #pragma unroll
        for (int mt = 0; mt < 2; ++mt)
            afr[mt] = ldsFrag(&As[cur][(wm * 32 + mt * 16 + (lane & 15)) * LDSS], lane);

        #pragma unroll
        for (int g = 0; g < 4; ++g) {
            v16bf bfr[2];
            #pragma unroll
            for (int nt = 0; nt < 2; ++nt)
                bfr[nt] = ldsFrag(&Bs[cur][(g * 64 + wn * 32 + nt * 16 + (lane & 15)) * LDSS], lane);
            #pragma unroll
            for (int mt = 0; mt < 2; ++mt)
                #pragma unroll
                for (int nt = 0; nt < 2; ++nt)
                    acc[g][mt][nt] = __builtin_amdgcn_wmma_f32_16x16x32_bf16(
                        false, afr[mt], false, bfr[nt],
                        (short)0, acc[g][mt][nt], false, false);
        }

        if (kc + 1 < NKC) storeLDS(cur ^ 1);
        cur ^= 1;
    }

    // Epilogue: biases + LSTM nonlinearities, fused per accumulator element.
    // C/D layout: lanes 0-15 -> (M = e, N = lane); lanes 16-31 -> (M = 8+e, N = lane-16).
    float* hyOut = out;
    float* cyOut = out + (size_t)B_DIM * H_DIM;
    const int nlo = lane & 15;
    const int mhi = (lane & 16) ? 8 : 0;
    #pragma unroll
    for (int mt = 0; mt < 2; ++mt) {
        #pragma unroll
        for (int nt = 0; nt < 2; ++nt) {
            const int col = n0 + wn * 32 + nt * 16 + nlo;
            const float bi = b_xh[0 * H_DIM + col] + b_hh[0 * H_DIM + col];
            const float bf = b_xh[1 * H_DIM + col] + b_hh[1 * H_DIM + col];
            const float bg = b_xh[2 * H_DIM + col] + b_hh[2 * H_DIM + col];
            const float bo = b_xh[3 * H_DIM + col] + b_hh[3 * H_DIM + col];
            const int mbase = m0 + wm * 32 + mt * 16 + mhi;
            #pragma unroll
            for (int e = 0; e < 8; ++e) {
                const int m = mbase + e;
                const float iv = fast_sigmoid(acc[0][mt][nt][e] + bi);
                const float fv = fast_sigmoid(acc[1][mt][nt][e] + bf);
                const float gv = fast_tanh  (acc[2][mt][nt][e] + bg);
                const float ov = fast_sigmoid(acc[3][mt][nt][e] + bo);
                const float cxv = cx[(size_t)m * H_DIM + col];
                const float cyv = cxv * fv + iv * gv;
                const float hyv = ov * fast_tanh(cyv);
                hyOut[(size_t)m * H_DIM + col] = hyv;
                cyOut[(size_t)m * H_DIM + col] = cyv;
            }
        }
    }
}

extern "C" void kernel_launch(void* const* d_in, const int* in_sizes, int n_in,
                              void* d_out, int out_size, void* d_ws, size_t ws_size,
                              hipStream_t stream) {
    (void)in_sizes; (void)n_in; (void)out_size; (void)d_ws; (void)ws_size;
    const float* x    = (const float*)d_in[0];
    const float* hx   = (const float*)d_in[1];
    const float* cx   = (const float*)d_in[2];
    const float* W_xh = (const float*)d_in[3];
    const float* b_xh = (const float*)d_in[4];
    const float* W_hh = (const float*)d_in[5];
    const float* b_hh = (const float*)d_in[6];
    float* out = (float*)d_out;

    dim3 grid(H_DIM / 64, B_DIM / 128);   // 16 x 64 = 1024 workgroups
    dim3 block(256);                      // 8 waves (wave32)
    hipLaunchKernelGGL(lstm_wmma_kernel, grid, block, 0, stream,
                       x, hx, cx, W_xh, b_xh, W_hh, b_hh, out);
}
